// GraphLayer_57732950393383
// MI455X (gfx1250) — matrix operand order repaired and checked by reading, not compiled, not run on hardware
//
#include <hip/hip_runtime.h>
#include <math.h>

// ============================================================================
// MI455X (gfx1250, wave32) graph-attention layer.
// GEMMs use v_wmma_f32_16x16x32_f16; B-fragments are staged transposed in LDS
// so every fragment load is a contiguous 32-byte ds_load (2x b128).
// Double segment-sum fused into one atomic scatter (linearity of sums).
// ============================================================================

typedef __attribute__((ext_vector_type(16))) _Float16 v16h;
typedef __attribute__((ext_vector_type(8)))  _Float16 v8h;
typedef __attribute__((ext_vector_type(8)))  float    v8f;

constexpr int B  = 4;
constexpr int N  = 2048;
constexpr int E  = 65536;
constexpr int F  = 128;
constexpr int FE = 64;
constexpr int H  = 8;
constexpr int DH = F / H;          // 16
constexpr int KX = 2 * F + FE;     // 320  (message GEMM K)
constexpr int K2 = 2 * F;          // 256  (qkv GEMM K)

// ---- output layout (flat floats, tuple order) ----
constexpr size_t OUT_UPD   = 0;
constexpr size_t OUT_WM    = (size_t)B * N * F;
constexpr size_t OUT_EDGES = OUT_WM + (size_t)B * E * F;
constexpr size_t OUT_EW    = OUT_EDGES + (size_t)B * E * 2;

// ---- workspace layout (bytes) ----
constexpr size_t AGG_OFF = 0;                                    // B*N*F f32
constexpr size_t Q16_OFF = AGG_OFF + (size_t)B * N * F * 4;      // [bh][n][dh]
constexpr size_t K16_OFF = Q16_OFF + (size_t)B * N * F * 2;      // [bh][n][dh]
constexpr size_t VT_OFF  = K16_OFF + (size_t)B * N * F * 2;      // [bh][dh][n]
constexpr size_t ATT_OFF = VT_OFF  + (size_t)B * N * F * 2;      // B*N*F f32

// ----------------------------------------------------------------------------
// Fragment builders (16x16x32 f16), this file's own contiguous k order (not the ISA table order).
//   A (16x32): lane&15 = M; lanes 0-15 K=0-15, lanes 16-31 K=16-31
//              (contiguous, matching the B fragment order).
//   B (32x16): lanes 0-15 K=0-15, lanes 16-31 K=16-31; lane&15 = N.
//              From a K-contiguous transposed buffer this is one 32B read.
//   C/D: VGPR v -> row v (lanes 0-15) / v+8 (lanes 16-31); lane&15 = N.
// ----------------------------------------------------------------------------
__device__ inline v16h frag_A(const _Float16* src, int ld) {
  const int lane = threadIdx.x & 31;
  const int m    = lane & 15;
  const int kb   = (lane >> 4) << 4;
  const _Float16* row = src + m * ld;
  v16h a;
#pragma unroll
  for (int j = 0; j < 8; ++j) a[j]     = row[kb + j];
#pragma unroll
  for (int j = 0; j < 8; ++j) a[8 + j] = row[kb + 8 + j];
  return a;
}

// Transposed-source B fragment: srcT is [Ncols][Kld] (K contiguous per row).
__device__ inline v16h frag_B_T(const _Float16* srcT, int ldT) {
  const int lane = threadIdx.x & 31;
  const int n    = lane & 15;
  const int kb   = (lane >> 4) << 4;       // 0 or 16
  return *(const v16h*)(srcT + n * ldT + kb);
}

__device__ inline v8f wmma_f16(v16h a, v16h b, v8f c) {
  return __builtin_amdgcn_wmma_f32_16x16x32_f16(false, a, false, b,
                                                (short)0, c, false, false);
}

__device__ inline float gelu_exact(float x) {
  return 0.5f * x * (1.0f + erff(x * 0.70710678118654752f));
}

__device__ inline float hsum16(float x) {
#pragma unroll
  for (int m = 8; m; m >>= 1) x += __shfl_xor(x, m);
  return x;
}
__device__ inline float hmax16(float x) {
#pragma unroll
  for (int m = 8; m; m >>= 1) x = fmaxf(x, __shfl_xor(x, m));
  return x;
}
__device__ inline float hsum8(float x) {
#pragma unroll
  for (int m = 4; m; m >>= 1) x += __shfl_xor(x, m);
  return x;
}

// ============================================================================
__global__ void zero_f32(float* p, int n) {
  int i = blockIdx.x * blockDim.x + threadIdx.x;
  if (i < n) p[i] = 0.0f;
}

// ============================================================================
// Kernel 1: message MLP.  32 edges x 128 outputs per block (8 waves, two
// M-accumulators each), K=320.  Epilogue: +bias, GELU, LayerNorm, * weight,
// write weighted_messages, atomic scatter of the fused double segment-sum.
// ============================================================================
__global__ void msg_kernel(const float* __restrict__ nodes,
                           const float* __restrict__ edge_features,
                           const int*   __restrict__ edges,
                           const float* __restrict__ edge_weights,
                           const float* __restrict__ W_msg,
                           const float* __restrict__ b_msg,
                           const float* __restrict__ g1,
                           const float* __restrict__ beta1,
                           float* __restrict__ out_wm,
                           float* __restrict__ agg) {
  const int tile = blockIdx.x;             // B*E/32 tiles
  const int b    = tile / (E / 32);
  const int e0   = (tile % (E / 32)) * 32;
  const int tid  = threadIdx.x;
  const int wave = tid >> 5;

  __shared__ alignas(16) _Float16 xbuf[32][KX];    // gathered features
  __shared__ alignas(64) _Float16 wbufT[F][32];    // W slab, K-contiguous
  __shared__ float obuf[32][F];

  for (int i = tid; i < 32 * KX; i += 256) {
    const int r = i / KX, c = i % KX;
    const int e = e0 + r;
    float val;
    if (c < F) {
      const int s = edges[((size_t)b * E + e) * 2 + 0];
      val = nodes[((size_t)b * N + s) * F + c];
    } else if (c < 2 * F) {
      const int d = edges[((size_t)b * E + e) * 2 + 1];
      val = nodes[((size_t)b * N + d) * F + (c - F)];
    } else {
      val = edge_features[((size_t)b * E + e) * FE + (c - 2 * F)];
    }
    xbuf[r][c] = (_Float16)val;
  }
  __syncthreads();

  v8f acc0 = {}, acc1 = {};
  for (int kk = 0; kk < KX; kk += 32) {
    __syncthreads();
    for (int i = tid; i < 32 * F; i += 256) {
      const int r = i >> 7, c = i & (F - 1);       // global-coalesced read
      wbufT[c][r] = (_Float16)W_msg[(size_t)(kk + r) * F + c];
    }
    if (kk + 32 < KX) __builtin_prefetch(&W_msg[(size_t)(kk + 32) * F], 0, 0);
    __syncthreads();
    v16h bf = frag_B_T(&wbufT[wave * 16][0], 32);
    v16h a0 = frag_A(&xbuf[0][0]  + kk, KX);
    v16h a1 = frag_A(&xbuf[16][0] + kk, KX);
    acc0 = wmma_f16(a0, bf, acc0);
    acc1 = wmma_f16(a1, bf, acc1);
  }

  {
    const int lane = tid & 31;
    const int half = lane >> 4;
    const int n    = wave * 16 + (lane & 15);
    const float bn = b_msg[n];
#pragma unroll
    for (int v = 0; v < 8; ++v) {
      obuf[v + 8 * half][n]      = acc0[v] + bn;
      obuf[16 + v + 8 * half][n] = acc1[v] + bn;
    }
  }
  __syncthreads();

  // 8 threads per edge row; 16 columns each.
  const int r   = tid >> 3;
  const int sub = tid & 7;
  const int e   = e0 + r;
  float ge[16];
  float sum = 0.0f;
#pragma unroll
  for (int q = 0; q < 16; ++q) {
    const float g = gelu_exact(obuf[r][sub + 8 * q]);
    ge[q] = g;
    sum += g;
  }
  const float mu = hsum8(sum) * (1.0f / F);
  float var = 0.0f;
#pragma unroll
  for (int q = 0; q < 16; ++q) {
    const float d = ge[q] - mu;
    var += d * d;
  }
  var = hsum8(var) * (1.0f / F);
  const float rs = rsqrtf(var + 1e-3f);

  const float w  = edge_weights[((size_t)b * E + e) * 2 + 1];
  const int  seg = (int)edge_weights[((size_t)b * E + e) * 2 + 0];
  const int dest = edges[((size_t)b * E + seg) * 2 + 1];
#pragma unroll
  for (int q = 0; q < 16; ++q) {
    const int c  = sub + 8 * q;
    const float m  = (ge[q] - mu) * rs * g1[c] + beta1[c];
    const float wm = m * w;
    out_wm[((size_t)b * E + e) * F + c] = wm;
    atomicAdd(&agg[((size_t)b * N + dest) * F + c], wm);
  }
}

// ============================================================================
// Kernel 2: QKV projections, 32 nodes per block, K=256.
//  q -> [bh][n][dh] f16, pre-scaled by 1/sqrt(DH)
//  k -> [bh][n][dh] f16
//  v -> [bh][dh][n] f16 (transposed, PV-fragment-ready)
// ============================================================================
__global__ void qkv_kernel(const float* __restrict__ nodes,
                           const float* __restrict__ agg,
                           const float* __restrict__ Wq, const float* __restrict__ bq,
                           const float* __restrict__ Wk, const float* __restrict__ bk,
                           const float* __restrict__ Wv, const float* __restrict__ bv,
                           _Float16* __restrict__ q16,
                           _Float16* __restrict__ k16,
                           _Float16* __restrict__ vT16) {
  const int tile = blockIdx.x;             // B*N/32 tiles
  const int b    = tile / (N / 32);
  const int n0   = (tile % (N / 32)) * 32;
  const int tid  = threadIdx.x;
  const int wave = tid >> 5;
  const int which = blockIdx.y;

  const float* W   = (which == 0) ? Wq : (which == 1) ? Wk : Wv;
  const float* bia = (which == 0) ? bq : (which == 1) ? bk : bv;

  __shared__ alignas(16) _Float16 xbuf[32][K2];
  __shared__ alignas(64) _Float16 wbufT[F][32];

  for (int i = tid; i < 32 * K2; i += 256) {
    const int r = i / K2, c = i % K2;
    const float val = (c < F) ? nodes[((size_t)b * N + n0 + r) * F + c]
                              : agg[((size_t)b * N + n0 + r) * F + (c - F)];
    xbuf[r][c] = (_Float16)val;
  }
  __syncthreads();

  v8f acc0 = {}, acc1 = {};
  for (int kk = 0; kk < K2; kk += 32) {
    __syncthreads();
    for (int i = tid; i < 32 * F; i += 256) {
      const int r = i >> 7, c = i & (F - 1);
      wbufT[c][r] = (_Float16)W[(size_t)(kk + r) * F + c];
    }
    __syncthreads();
    v16h bf = frag_B_T(&wbufT[wave * 16][0], 32);
    v16h a0 = frag_A(&xbuf[0][0]  + kk, K2);
    v16h a1 = frag_A(&xbuf[16][0] + kk, K2);
    acc0 = wmma_f16(a0, bf, acc0);
    acc1 = wmma_f16(a1, bf, acc1);
  }

  const int lane = tid & 31;
  const int half = lane >> 4;
  const int n    = wave * 16 + (lane & 15);
  const int h    = n >> 4, d = n & 15;
  const float bn = bia[n];
  const float sc = (which == 0) ? 0.25f : 1.0f;    // fold 1/sqrt(DH) into q
  const size_t bh = (size_t)(b * H + h);
#pragma unroll
  for (int v = 0; v < 8; ++v) {
    const int node0 = n0 + v + 8 * half;
    const int node1 = node0 + 16;
    const float y0 = (acc0[v] + bn) * sc;
    const float y1 = (acc1[v] + bn) * sc;
    if (which == 2) {
      vT16[(bh * DH + d) * N + node0] = (_Float16)y0;
      vT16[(bh * DH + d) * N + node1] = (_Float16)y1;
    } else {
      _Float16* outp = (which == 0) ? q16 : k16;
      outp[(bh * N + node0) * DH + d] = (_Float16)y0;
      outp[(bh * N + node1) * DH + d] = (_Float16)y1;
    }
  }
}

// ============================================================================
// Kernel 3: flash attention.  8 waves x 16-query tiles per block (128 queries)
// over one (b,h).  Per 32-key tile: 2 score WMMAs (DH padded 16->32), online
// softmax, 1 PV WMMA.  Keys within a tile are interleaved (pos 2n <- key n,
// pos 2n+1 <- key n+16) so P can be written as packed b32; V staging applies
// the identical permutation (all 32 keys of a tile share one 64B line).
// ============================================================================
__global__ void attn_kernel(const _Float16* __restrict__ q16,
                            const _Float16* __restrict__ k16,
                            const _Float16* __restrict__ vT16,
                            float* __restrict__ attout) {
  constexpr int QCH = N / 128;
  const int bh   = blockIdx.x / QCH;
  const int qc   = blockIdx.x % QCH;
  const int tid  = threadIdx.x;            // 256 threads, 8 waves
  const int wave = tid >> 5;
  const int lane = tid & 31;
  const int q0   = qc * 128 + wave * 16;

  const _Float16* qb = q16  + (size_t)bh * N * DH;
  const _Float16* kb = k16  + (size_t)bh * N * DH;
  const _Float16* vT = vT16 + (size_t)bh * DH * N;

  __shared__ alignas(32) _Float16 kbuf[32][DH];    // [key][dh]
  __shared__ alignas(64) _Float16 vTbuf[DH][32];   // [dh][key-pos], permuted
  __shared__ alignas(16) _Float16 pbuf[8][16][32]; // per-wave P tiles

  // Q A-fragment: lanes 0-15 hold dims 0-15 (pre-scaled), lanes 16-31 (K 16-31) zero-padded.
  v16h aq = {};
  {
    const int m  = lane & 15;
    if (lane < 16) aq = *(const v16h*)(qb + (size_t)(q0 + m) * DH);
  }

  v8f o = {};
  float mi[8], li[8];
#pragma unroll
  for (int v = 0; v < 8; ++v) { mi[v] = -1e30f; li[v] = 0.0f; }

  const int half = lane >> 4;
  const int nn   = lane & 15;

  for (int kt = 0; kt < N / 32; ++kt) {
    __syncthreads();
    for (int i = tid; i < 32 * DH; i += 256) {
      const int r = i >> 4, c = i & 15;
      kbuf[r][c] = kb[(size_t)(kt * 32 + r) * DH + c];
    }
    for (int i = tid; i < DH * 32; i += 256) {
      const int c = i >> 5, pos = i & 31;
      const int key = (pos & 1) ? 16 + (pos >> 1) : (pos >> 1);
      vTbuf[c][pos] = vT[(size_t)c * N + kt * 32 + key];
    }
    __syncthreads();

    // Score fragments: lanes 0-15 read one contiguous K-row, lanes 16-31 are
    // the zero-padded dims.
    v8f s0 = {}, s1 = {};
    {
      v16h bf0 = {}, bf1 = {};
      if (lane < 16) {
        bf0 = *(const v16h*)&kbuf[nn][0];
        bf1 = *(const v16h*)&kbuf[16 + nn][0];
      }
      s0 = wmma_f16(aq, bf0, s0);
      s1 = wmma_f16(aq, bf1, s1);
    }

    // Online softmax per row (row = v + 8*half).
#pragma unroll
    for (int v = 0; v < 8; ++v) {
      const float mx = hmax16(fmaxf(s0[v], s1[v]));
      const float mn = fmaxf(mi[v], mx);
      const float corr = __expf(mi[v] - mn);
      const float p0 = __expf(s0[v] - mn);
      const float p1 = __expf(s1[v] - mn);
      li[v] = li[v] * corr + hsum16(p0 + p1);
      o[v] *= corr;
      mi[v] = mn;
      union { _Float16 h[2]; unsigned u; } pk;
      pk.h[0] = (_Float16)p0;                      // key-pos 2n   (key n)
      pk.h[1] = (_Float16)p1;                      // key-pos 2n+1 (key n+16)
      *(unsigned*)&pbuf[wave][v + 8 * half][2 * nn] = pk.u;
    }
    __syncthreads();

    v16h ap  = frag_A(&pbuf[wave][0][0], 32);
    v16h bfv = frag_B_T(&vTbuf[0][0], 32);
    o = wmma_f16(ap, bfv, o);
  }

  const int b = bh >> 3;
  const int h = bh & 7;
#pragma unroll
  for (int v = 0; v < 8; ++v) {
    const int node = q0 + v + 8 * half;
    attout[((size_t)b * N + node) * F + h * DH + nn] = o[v] / li[v];
  }
}

// ============================================================================
// Kernel 4: output projection (32 nodes/block, K=128) + GELU + LayerNorm.
// ============================================================================
__global__ void outproj_kernel(const float* __restrict__ att,
                               const float* __restrict__ Wo,
                               const float* __restrict__ bo,
                               const float* __restrict__ g2,
                               const float* __restrict__ beta2,
                               float* __restrict__ out_upd) {
  const int tile = blockIdx.x;             // B*N/32
  const int b    = tile / (N / 32);
  const int n0   = (tile % (N / 32)) * 32;
  const int tid  = threadIdx.x;
  const int wave = tid >> 5;

  __shared__ alignas(16) _Float16 xbuf[32][F];
  __shared__ alignas(64) _Float16 wbufT[F][32];
  __shared__ float obuf[32][F];

  for (int i = tid; i < 32 * F; i += 256) {
    const int r = i >> 7, c = i & (F - 1);
    xbuf[r][c] = (_Float16)att[((size_t)b * N + n0 + r) * F + c];
  }
  __syncthreads();

  v8f acc0 = {}, acc1 = {};
  for (int kk = 0; kk < F; kk += 32) {
    __syncthreads();
    for (int i = tid; i < 32 * F; i += 256) {
      const int r = i >> 7, c = i & (F - 1);
      wbufT[c][r] = (_Float16)Wo[(size_t)(kk + r) * F + c];
    }
    __syncthreads();
    v16h bf = frag_B_T(&wbufT[wave * 16][0], 32);
    v16h a0 = frag_A(&xbuf[0][0]  + kk, F);
    v16h a1 = frag_A(&xbuf[16][0] + kk, F);
    acc0 = wmma_f16(a0, bf, acc0);
    acc1 = wmma_f16(a1, bf, acc1);
  }

  {
    const int lane = tid & 31;
    const int half = lane >> 4;
    const int n    = wave * 16 + (lane & 15);
    const float bn = bo[n];
#pragma unroll
    for (int v = 0; v < 8; ++v) {
      obuf[v + 8 * half][n]      = acc0[v] + bn;
      obuf[16 + v + 8 * half][n] = acc1[v] + bn;
    }
  }
  __syncthreads();

  const int r   = tid >> 3;
  const int sub = tid & 7;
  float ge[16];
  float sum = 0.0f;
#pragma unroll
  for (int q = 0; q < 16; ++q) {
    const float g = gelu_exact(obuf[r][sub + 8 * q]);
    ge[q] = g;
    sum += g;
  }
  const float mu = hsum8(sum) * (1.0f / F);
  float var = 0.0f;
#pragma unroll
  for (int q = 0; q < 16; ++q) {
    const float d = ge[q] - mu;
    var += d * d;
  }
  var = hsum8(var) * (1.0f / F);
  const float rs = rsqrtf(var + 1e-3f);
#pragma unroll
  for (int q = 0; q < 16; ++q) {
    const int c = sub + 8 * q;
    out_upd[((size_t)b * N + n0 + r) * F + c] =
        (ge[q] - mu) * rs * g2[c] + beta2[c];
  }
}

// ============================================================================
__global__ void copy_i2f(const int* __restrict__ src, float* __restrict__ dst, int n) {
  int i = blockIdx.x * blockDim.x + threadIdx.x;
  if (i < n) dst[i] = (float)src[i];
}
__global__ void copy_f(const float* __restrict__ src, float* __restrict__ dst, int n) {
  int i = blockIdx.x * blockDim.x + threadIdx.x;
  if (i < n) dst[i] = src[i];
}

// ============================================================================
extern "C" void kernel_launch(void* const* d_in, const int* in_sizes, int n_in,
                              void* d_out, int out_size, void* d_ws, size_t ws_size,
                              hipStream_t stream) {
  const float* nodes         = (const float*)d_in[0];
  const float* edge_features = (const float*)d_in[1];
  const int*   edges         = (const int*)  d_in[2];
  const float* edge_weights  = (const float*)d_in[3];
  const float* W_msg = (const float*)d_in[4];
  const float* b_msg = (const float*)d_in[5];
  const float* g1    = (const float*)d_in[6];
  const float* beta1 = (const float*)d_in[7];
  const float* Wq = (const float*)d_in[8];  const float* bq = (const float*)d_in[9];
  const float* Wk = (const float*)d_in[10]; const float* bk = (const float*)d_in[11];
  const float* Wv = (const float*)d_in[12]; const float* bv = (const float*)d_in[13];
  const float* Wo = (const float*)d_in[14]; const float* bo = (const float*)d_in[15];
  const float* g2 = (const float*)d_in[16]; const float* beta2 = (const float*)d_in[17];

  float* out = (float*)d_out;
  char*  ws  = (char*)d_ws;
  float*     agg  = (float*)    (ws + AGG_OFF);
  _Float16*  q16  = (_Float16*) (ws + Q16_OFF);
  _Float16*  k16  = (_Float16*) (ws + K16_OFF);
  _Float16*  vT16 = (_Float16*) (ws + VT_OFF);
  float*     attb = (float*)    (ws + ATT_OFF);

  zero_f32<<<(B * N * F + 255) / 256, 256, 0, stream>>>(agg, B * N * F);

  msg_kernel<<<B * E / 32, 256, 0, stream>>>(
      nodes, edge_features, edges, edge_weights, W_msg, b_msg, g1, beta1,
      out + OUT_WM, agg);

  dim3 gq(B * N / 32, 3);
  qkv_kernel<<<gq, 256, 0, stream>>>(nodes, agg, Wq, bq, Wk, bk, Wv, bv,
                                     q16, k16, vT16);

  attn_kernel<<<B * H * (N / 128), 256, 0, stream>>>(q16, k16, vT16, attb);

  outproj_kernel<<<B * N / 32, 256, 0, stream>>>(attb, Wo, bo, g2, beta2,
                                                 out + OUT_UPD);

  copy_i2f<<<(B * E * 2 + 255) / 256, 256, 0, stream>>>(edges, out + OUT_EDGES,
                                                        B * E * 2);
  copy_f<<<(B * E * 2 + 255) / 256, 256, 0, stream>>>(edge_weights,
                                                      out + OUT_EW, B * E * 2);
}
